// StackedEncoder_43568148250639
// MI455X (gfx1250) — matrix operand order repaired and checked by reading, not compiled, not run on hardware
//
#include <hip/hip_runtime.h>

typedef _Float16 v16h __attribute__((ext_vector_type(16)));
typedef _Float16 v8h  __attribute__((ext_vector_type(8)));
typedef float    v8f  __attribute__((ext_vector_type(8)));

static constexpr int FEAT = 64;

// ---------------- utility kernels ----------------

__global__ void zero_f32(float* __restrict__ p, int n) {
    int i = blockIdx.x * blockDim.x + threadIdx.x;
    if (i < n) p[i] = 0.0f;
}

__global__ void deg_kernel(const int* __restrict__ dst, float* __restrict__ deg, int E) {
    int e = blockIdx.x * blockDim.x + threadIdx.x;
    if (e < E) unsafeAtomicAdd(&deg[dst[e]], 1.0f);
}

__global__ void invdeg_kernel(float* __restrict__ deg, int N) {
    int i = blockIdx.x * blockDim.x + threadIdx.x;
    if (i < N) {
        float d = deg[i];
        deg[i] = d > 0.0f ? 1.0f / d : 0.0f;
    }
}

// Build B^T in f16: BmT[l][c][k], c in [0,128) output col (0..63 = u gate, 64..127 = c gate),
// k in [0,128) reduction (0..63 from agg_x weights, 64..127 from agg_h weights).
// gate order in W: 0=r_x 1=r_h 2=u_x 3=u_h 4=c_x 5=c_h  (r is faithfully unused)
__global__ void convert_w(const float* __restrict__ W, _Float16* __restrict__ BmT, int total) {
    int tid = blockIdx.x * blockDim.x + threadIdx.x;
    if (tid >= total) return;
    int l   = tid >> 14;
    int rem = tid & 16383;
    int c   = rem >> 7;
    int k   = rem & 127;
    int cc  = c & 63;
    int g, row;
    if (k < 64) { g = (c < 64) ? 2 : 4; row = k; }
    else        { g = (c < 64) ? 3 : 5; row = k - 64; }
    BmT[tid] = (_Float16)W[((l * 6 + g) * FEAT + row) * FEAT + cc];
}

// Gather features of both streams along edges and scatter-add into agg buffers.
// One thread = (edge, 2 feats). Mean-scaling (inv_deg) is folded into the GEMM A-load.
__global__ void edge_agg(const float* __restrict__ inp, const float* __restrict__ hcur,
                         const int* __restrict__ src, const int* __restrict__ dst,
                         float* __restrict__ agg_x, float* __restrict__ agg_h, int E) {
    int t = blockIdx.x * blockDim.x + threadIdx.x;
    if (t >= E * 32) return;
    int e = t >> 5;
    int f = (t & 31) << 1;
    int s = src[e] * FEAT + f;
    int d = dst[e] * FEAT + f;
    float2 vx = *(const float2*)(inp + s);
    float2 vh = *(const float2*)(hcur + s);
    unsafeAtomicAdd(&agg_x[d],     vx.x);
    unsafeAtomicAdd(&agg_x[d + 1], vx.y);
    unsafeAtomicAdd(&agg_h[d],     vh.x);
    unsafeAtomicAdd(&agg_h[d + 1], vh.y);
}

// ---------------- fused WMMA GEMM + GRU gate kernel ----------------
// A = [agg_x*inv_deg | agg_h*inv_deg]  (N x 128, f32->f16 on the fly)
// B^T staged in LDS (128x128 f16 = 32KB).  Each wave: 16 rows x 128 cols.
// cols 0..63 -> u (sigmoid), 64..127 -> c (tanh);  h = u*h + (1-u)*c  in place.
__global__ __launch_bounds__(256) void gemm_gate(
    const float* __restrict__ agg_x, const float* __restrict__ agg_h,
    const float* __restrict__ inv_deg, const _Float16* __restrict__ BmT,
    const float* __restrict__ bias /* b[l] : [6][64] */,
    float* __restrict__ hbuf, int N)
{
    __shared__ uint4 sBraw[2048]; // 128*128 f16
    const uint4* gB = (const uint4*)BmT;
    for (int i = threadIdx.x; i < 2048; i += 256) sBraw[i] = gB[i];
    __syncthreads();
    const _Float16* sB = (const _Float16*)sBraw;

    const int lane = threadIdx.x & 31;
    const int wave = threadIdx.x >> 5;
    const int m    = lane & 15;   // row within A frag / col within B,C,D frag
    const int half = lane >> 4;   // K-half select (A/B), row-block select (C/D)
    const int rowBase = blockIdx.x * 128 + wave * 16;
    const int r  = rowBase + m;
    const int rc = (r < N) ? r : (N - 1);
    const float s = (r < N) ? inv_deg[r] : 0.0f;

    v8f zero = {0.f, 0.f, 0.f, 0.f, 0.f, 0.f, 0.f, 0.f};
    v8f acc[8];
#pragma unroll
    for (int i = 0; i < 8; ++i) acc[i] = zero;

#pragma unroll
    for (int ks = 0; ks < 4; ++ks) {
        const int kbase = ks * 32;
        const float* Ab = (kbase < 64) ? agg_x : agg_h;
        const int koff = (kbase & 63) + half * 8;
        // 16-bit A layout: lane holds K = koff+{0..7} and koff+16+{0..7} of its row
        const float4* ap = (const float4*)(Ab + rc * FEAT + koff);
        float4 f0 = ap[0], f1 = ap[1], f2 = ap[4], f3 = ap[5];
        float tmp[16] = {f0.x, f0.y, f0.z, f0.w, f1.x, f1.y, f1.z, f1.w,
                         f2.x, f2.y, f2.z, f2.w, f3.x, f3.y, f3.z, f3.w};
        v16h a;
#pragma unroll
        for (int i = 0; i < 16; ++i) a[i] = (_Float16)(tmp[i] * s);

#pragma unroll
        for (int cb = 0; cb < 8; ++cb) {
            const int n = cb * 16 + m;
            // 16-bit B layout: lane holds col n, K = kbase + half*8 + {0..7} and kbase + 16 + half*8 + {0..7}
            const v8h* bp = (const v8h*)(sB + n * 128 + kbase + half * 8);
            v8h blo = bp[0], bhi = bp[2];
            v16h bfrag;
#pragma unroll
            for (int i = 0; i < 8; ++i) { bfrag[i] = blo[i]; bfrag[8 + i] = bhi[i]; }
            acc[cb] = __builtin_amdgcn_wmma_f32_16x16x32_f16(
                false, a, false, bfrag, (short)0, acc[cb], false, false);
        }
    }

    // Epilogue: tiles 0..3 = u-gate preact, tiles 4..7 = c-gate preact (same lane/row).
#pragma unroll
    for (int cb = 0; cb < 4; ++cb) {
        const int col = cb * 16 + m;
        const float bu = bias[2 * FEAT + col] + bias[3 * FEAT + col];
        const float bc = bias[4 * FEAT + col] + bias[5 * FEAT + col];
#pragma unroll
        for (int j = 0; j < 8; ++j) {
            const int row = rowBase + half * 8 + j;
            if (row < N) {
                float up = acc[cb][j] + bu;
                float cp = acc[cb + 4][j] + bc;
                float u  = 1.0f / (1.0f + __expf(-up));
                float e2 = __expf(2.0f * cp);
                float cg = (e2 - 1.0f) / (e2 + 1.0f);
                float ho = hbuf[row * FEAT + col];
                hbuf[row * FEAT + col] = u * ho + (1.0f - u) * cg;
            }
        }
    }
}

// ---------------- host orchestration ----------------

extern "C" void kernel_launch(void* const* d_in, const int* in_sizes, int n_in,
                              void* d_out, int out_size, void* d_ws, size_t ws_size,
                              hipStream_t stream) {
    const float* x  = (const float*)d_in[0];
    const float* h0 = (const float*)d_in[1];
    const float* W  = (const float*)d_in[2];
    const float* b  = (const float*)d_in[3];
    const int* src  = (const int*)d_in[4];
    const int* dst  = (const int*)d_in[5];

    const int L = in_sizes[2] / (6 * FEAT * FEAT);
    const int N = in_sizes[1] / (L * FEAT);
    const int T = in_sizes[0] / (N * FEAT);
    const int E = in_sizes[4];

    float* out  = (float*)d_out;
    float* hbuf = out + (size_t)T * N * FEAT;   // h lives in d_out's h-region

    // workspace layout
    float* inv_deg = (float*)d_ws;
    size_t nAlign = (size_t)((N + 63) & ~63);
    float* agg_x = inv_deg + nAlign;
    float* agg_h = agg_x + (size_t)N * FEAT;
    _Float16* BmT = (_Float16*)(agg_h + (size_t)N * FEAT);

    // x passthrough + h0 init (both fully overwrite their d_out regions each call)
    hipMemcpyAsync(out,  x,  (size_t)T * N * FEAT * sizeof(float),
                   hipMemcpyDeviceToDevice, stream);
    hipMemcpyAsync(hbuf, h0, (size_t)L * N * FEAT * sizeof(float),
                   hipMemcpyDeviceToDevice, stream);

    // in-degree -> inv_deg (recomputed every call; deterministic)
    zero_f32<<<(N + 255) / 256, 256, 0, stream>>>(inv_deg, N);
    deg_kernel<<<(E + 255) / 256, 256, 0, stream>>>(dst, inv_deg, E);
    invdeg_kernel<<<(N + 255) / 256, 256, 0, stream>>>(inv_deg, N);

    // f16 B^T weights
    const int wtot = L * 128 * 128;
    convert_w<<<(wtot + 255) / 256, 256, 0, stream>>>(W, BmT, wtot);

    const int aggN  = 2 * N * FEAT;
    const int eThr  = E * 32;
    const int gemmB = (N + 127) / 128;

    for (int t = 0; t < T; ++t) {
        for (int l = 0; l < L; ++l) {
            const float* inp  = (l == 0) ? (x + (size_t)t * N * FEAT)
                                         : (hbuf + (size_t)(l - 1) * N * FEAT);
            float* hcur = hbuf + (size_t)l * N * FEAT;

            zero_f32<<<(aggN + 255) / 256, 256, 0, stream>>>(agg_x, aggN);
            edge_agg<<<(eThr + 255) / 256, 256, 0, stream>>>(
                inp, hcur, src, dst, agg_x, agg_h, E);
            gemm_gate<<<gemmB, 256, 0, stream>>>(
                agg_x, agg_h, inv_deg,
                BmT + (size_t)l * 128 * 128,
                b + (size_t)l * 6 * FEAT,
                hcur, N);
        }
    }
}
